// Starcoder2Attention_66236985639141
// MI455X (gfx1250) — matrix operand order repaired and checked by reading, not compiled, not run on hardware
//
#include <hip/hip_runtime.h>
#include <hip/hip_bf16.h>
#include <stddef.h>

// ---------------------------------------------------------------------------
// Problem constants (Starcoder2 attention)
// ---------------------------------------------------------------------------
#define BB   2
#define SS   2048
#define HH   24
#define KK   2
#define DD   128
#define WIN  1024
#define HID  (HH * DD)            // 3072
#define TT   (BB * SS)            // 4096
#define QKVN ((HH + 2 * KK) * DD) // 3584

typedef __attribute__((ext_vector_type(16))) _Float16 v16h;
typedef __attribute__((ext_vector_type(8)))  _Float16 v8h;
typedef __attribute__((ext_vector_type(8)))  float    v8f;

// ---------------------------------------------------------------------------
// WMMA helpers (gfx1250 v_wmma_f32_16x16x32_f16, wave32)
// A (16x32 f16): lane<16 -> M=lane, K halves {kc..kc+7, kc+16..kc+23}, kc=half*8
// B (32x16 f16): lane<16 -> N=lane, K halves {kc..kc+7, kc+16..kc+23}, kc=half*8
// C/D (16x16 f32): lane -> N=lane&15, VGPR r -> M = r + (lane>=16 ? 8 : 0)
// ---------------------------------------------------------------------------
__device__ __forceinline__ v8f wmma_f16(v16h a, v16h b, v8f c) {
  return __builtin_amdgcn_wmma_f32_16x16x32_f16(
      /*neg_a=*/false, a, /*neg_b=*/false, b,
      /*c_mod=*/(short)0, c, /*reuse_a=*/false, /*reuse_b=*/false);
}

__device__ __forceinline__ v16h load_a_frag(const _Float16* row, int kBase, int half) {
  const v8h lo = *(const v8h*)(row + kBase + half * 8);
  const v8h hi = *(const v8h*)(row + kBase + half * 8 + 16);
  v16h r;
#pragma unroll
  for (int i = 0; i < 8; ++i) { r[i] = lo[i]; r[i + 8] = hi[i]; }
  return r;
}

__device__ __forceinline__ v16h load_b_frag(const _Float16* col, int kBase, int half) {
  return load_a_frag(col, kBase, half);
}

// ---------------------------------------------------------------------------
// Kernel 1: f32 -> f16 conversion (grid-stride)
// ---------------------------------------------------------------------------
__global__ void cvt_f16_kernel(const float* __restrict__ src,
                               _Float16* __restrict__ dst, int n) {
  int stride = gridDim.x * blockDim.x;
  for (int i = blockIdx.x * blockDim.x + threadIdx.x; i < n; i += stride)
    dst[i] = (_Float16)src[i];
}

// ---------------------------------------------------------------------------
// Kernel 2/5: WMMA GEMM  out[m,n] = sum_k A[m,k] * W[n,k] + bias[n]
// Wave computes a 32(M) x 64(N) tile: each B fragment is reused for two
// M-subtiles (8 WMMAs per 2KB A + 4KB B per 32-k step -> ~22 FLOP/byte).
// M%32==0, N%64==0, K%32==0 required.
// ---------------------------------------------------------------------------
template <int OUT16>
__global__ __launch_bounds__(256) void gemm_wmma_kernel(
    const _Float16* __restrict__ A, const _Float16* __restrict__ W,
    const float* __restrict__ bias, _Float16* __restrict__ outH,
    float* __restrict__ outF, int M, int N, int Kd) {
  const int lane = threadIdx.x & 31;
  const int wid  = (blockIdx.x * blockDim.x + threadIdx.x) >> 5;
  const int mTiles  = M >> 5;
  const int nStrips = N >> 6;
  if (wid >= mTiles * nStrips) return;  // uniform per wave

  const int mT = wid % mTiles;
  const int nS = wid / mTiles;
  const int m0 = mT << 5;
  const int n0 = nS << 6;
  const int half = lane >> 4;
  const int lr   = lane & 15;

  const _Float16* arow0 = A + (size_t)(m0 + lr) * Kd;
  const _Float16* arow1 = A + (size_t)(m0 + 16 + lr) * Kd;

  v8f acc[8] = {};
  for (int k0 = 0; k0 < Kd; k0 += 32) {
    if (k0 + 64 < Kd) {  // hide the A stream behind the WMMA chain
      __builtin_prefetch(arow0 + k0 + 64, 0, 1);
      __builtin_prefetch(arow1 + k0 + 64, 0, 1);
    }
    v16h a0 = load_a_frag(arow0, k0, half);
    v16h a1 = load_a_frag(arow1, k0, half);
#pragma unroll
    for (int i = 0; i < 4; ++i) {
      v16h b = load_b_frag(W + (size_t)(n0 + i * 16 + lr) * Kd, k0, half);
      acc[i]     = wmma_f16(a0, b, acc[i]);
      acc[4 + i] = wmma_f16(a1, b, acc[4 + i]);
    }
  }

#pragma unroll
  for (int i = 0; i < 4; ++i) {
    const int n = n0 + i * 16 + lr;
    const float bn = bias[n];
#pragma unroll
    for (int r = 0; r < 8; ++r) {
      const int m = m0 + r + half * 8;
      const float v0 = acc[i][r] + bn;
      const float v1 = acc[4 + i][r] + bn;
      if (OUT16) {
        outH[(size_t)m * N + n]        = (_Float16)v0;
        outH[(size_t)(m + 16) * N + n] = (_Float16)v1;
      } else {
        outF[(size_t)m * N + n]        = v0;
        outF[(size_t)(m + 16) * N + n] = v1;
      }
    }
  }
}

// ---------------------------------------------------------------------------
// Kernel 3: RoPE (neox rotate-halves) + scatter into attention layouts.
//   qh[b][h][s][d], kh[b][kv][s][d], vT[b][kv][d][s]
// One thread per (t, head-slot, d<64) pair.
// ---------------------------------------------------------------------------
__global__ void rope_scatter_kernel(const _Float16* __restrict__ qkv,
                                    const float* __restrict__ cosT,
                                    const float* __restrict__ sinT,
                                    _Float16* __restrict__ qh,
                                    _Float16* __restrict__ kh,
                                    _Float16* __restrict__ vT) {
  const int total = TT * (HH + 2 * KK) * 64;
  int idx = blockIdx.x * blockDim.x + threadIdx.x;
  if (idx >= total) return;
  const int d   = idx & 63;
  int rem       = idx >> 6;
  const int hh  = rem % (HH + 2 * KK);
  const int t   = rem / (HH + 2 * KK);
  const int b   = t / SS;
  const int s   = t - b * SS;
  const _Float16* row = qkv + (size_t)t * QKVN;

  if (hh < HH + KK) {  // q or k head -> RoPE
    const float c  = cosT[t * 64 + d];
    const float sn = sinT[t * 64 + d];
    float x1, x2;
    _Float16* dst;
    if (hh < HH) {
      x1 = (float)row[hh * DD + d];
      x2 = (float)row[hh * DD + d + 64];
      dst = qh + ((size_t)(b * HH + hh) * SS + s) * DD;
    } else {
      const int kv = hh - HH;
      x1 = (float)row[HH * DD + kv * DD + d];
      x2 = (float)row[HH * DD + kv * DD + d + 64];
      dst = kh + ((size_t)(b * KK + kv) * SS + s) * DD;
    }
    dst[d]      = (_Float16)(x1 * c - x2 * sn);
    dst[d + 64] = (_Float16)(x2 * c + x1 * sn);
  } else {             // v head -> transpose, no RoPE
    const int kv = hh - HH - KK;
    const _Float16 x1 = row[(HH + KK) * DD + kv * DD + d];
    const _Float16 x2 = row[(HH + KK) * DD + kv * DD + d + 64];
    vT[((size_t)(b * KK + kv) * DD + d)      * SS + s] = x1;
    vT[((size_t)(b * KK + kv) * DD + d + 64) * SS + s] = x2;
  }
}

// ---------------------------------------------------------------------------
// Kernel 4: sliding-window causal flash attention.
// One wave per (b, h, 16-row q-tile). Online softmax; P re-shaped through LDS.
// ---------------------------------------------------------------------------
__global__ __launch_bounds__(256) void flash_attn_kernel(
    const _Float16* __restrict__ qh, const _Float16* __restrict__ kh,
    const _Float16* __restrict__ vT, _Float16* __restrict__ attnOut) {
  __shared__ alignas(16) _Float16 Pl[8][16 * 32];

  const int lane = threadIdx.x & 31;
  const int wIn  = threadIdx.x >> 5;
  const int wid  = blockIdx.x * 8 + wIn;
  const int QT   = SS / 16;                 // 128 q-tiles per head
  const int qt   = wid % QT;
  const int h    = (wid / QT) % HH;
  const int b    = wid / (QT * HH);
  const int kvh  = h / (HH / KK);
  const int q0   = qt * 16;
  const int half = lane >> 4;
  const int lr   = lane & 15;
  const float scale = 0.08838834764831845f;  // 1/sqrt(128)

  const _Float16* qrow = qh + ((size_t)((b * HH + h) * SS) + q0 + lr) * DD;
  v16h qa[4];
#pragma unroll
  for (int kk = 0; kk < 4; ++kk) qa[kk] = load_a_frag(qrow, kk * 32, half);

  const _Float16* kbase = kh + (size_t)((b * KK + kvh) * SS) * DD;
  const _Float16* vbase = vT + (size_t)((b * KK + kvh) * DD) * SS;

  float mrow[8], lrow[8];
  v8f acc[8] = {};
#pragma unroll
  for (int r = 0; r < 8; ++r) { mrow[r] = -1.0e30f; lrow[r] = 0.0f; }

  int kLo = q0 - WIN;
  if (kLo < 0) kLo = 0;
  kLo &= ~31;
  const int kHi = q0 + 15;
  _Float16* P = &Pl[wIn][0];

  for (int k0 = kLo; k0 <= kHi; k0 += 32) {
    // --- scores: two 16x16 tiles over k0..k0+31 ---
    v8f s[2] = {};
#pragma unroll
    for (int j = 0; j < 2; ++j) {
#pragma unroll
      for (int kk = 0; kk < 4; ++kk) {
        v16h kb = load_b_frag(kbase + (size_t)(k0 + j * 16 + lr) * DD, kk * 32, half);
        s[j] = wmma_f16(qa[kk], kb, s[j]);
      }
    }

    // --- mask + scale ---
    float sv[2][8];
#pragma unroll
    for (int j = 0; j < 2; ++j) {
      const int kpos = k0 + j * 16 + lr;
#pragma unroll
      for (int r = 0; r < 8; ++r) {
        const int qpos = q0 + r + half * 8;
        const bool ok  = (kpos <= qpos) && (qpos - kpos <= WIN);
        sv[j][r] = ok ? s[j][r] * scale : -1.0e30f;
      }
    }

    // --- online softmax per row (reduce across 16 lanes of the half-wave) ---
#pragma unroll
    for (int r = 0; r < 8; ++r) {
      float rm = fmaxf(sv[0][r], sv[1][r]);
#pragma unroll
      for (int off = 1; off < 16; off <<= 1)
        rm = fmaxf(rm, __shfl_xor(rm, off, 32));
      const float mnew = fmaxf(mrow[r], rm);
      const float fold = __expf(mrow[r] - mnew);
      mrow[r] = mnew;
      lrow[r] *= fold;
#pragma unroll
      for (int dt = 0; dt < 8; ++dt) acc[dt][r] *= fold;

      const float p0 = (sv[0][r] > -1.0e29f) ? __expf(sv[0][r] - mnew) : 0.0f;
      const float p1 = (sv[1][r] > -1.0e29f) ? __expf(sv[1][r] - mnew) : 0.0f;
      float rs = p0 + p1;
#pragma unroll
      for (int off = 1; off < 16; off <<= 1)
        rs += __shfl_xor(rs, off, 32);
      lrow[r] += rs;

      P[(r + half * 8) * 32 + lr]      = (_Float16)p0;
      P[(r + half * 8) * 32 + 16 + lr] = (_Float16)p1;
    }

    // --- P @ V: re-load P as A-fragment from LDS (ds_load_b128 x2) ---
    v16h pa = load_a_frag(P + lr * 32, 0, half);
#pragma unroll
    for (int dt = 0; dt < 8; ++dt) {
      v16h vb = load_b_frag(vbase + (size_t)(dt * 16 + lr) * SS, k0, half);
      acc[dt] = wmma_f16(pa, vb, acc[dt]);
    }
  }

  // --- normalize and store: attnOut[t][h*128 + d] ---
#pragma unroll
  for (int r = 0; r < 8; ++r) {
    const float inv = 1.0f / lrow[r];
    const int t = b * SS + q0 + r + half * 8;
    _Float16* orow = attnOut + (size_t)t * HID + h * DD;
#pragma unroll
    for (int dt = 0; dt < 8; ++dt)
      orow[dt * 16 + lr] = (_Float16)(acc[dt][r] * inv);
  }
}

// ---------------------------------------------------------------------------
// Host: launch sequence (all on `stream`, workspace carved from d_ws)
// ---------------------------------------------------------------------------
extern "C" void kernel_launch(void* const* d_in, const int* in_sizes, int n_in,
                              void* d_out, int out_size, void* d_ws, size_t ws_size,
                              hipStream_t stream) {
  const float* hidden = (const float*)d_in[0];
  const float* cosT   = (const float*)d_in[1];
  const float* sinT   = (const float*)d_in[2];
  const float* qkv_w  = (const float*)d_in[3];
  const float* qkv_b  = (const float*)d_in[4];
  const float* o_w    = (const float*)d_in[5];
  const float* o_b    = (const float*)d_in[6];
  float* out = (float*)d_out;

  // workspace layout (bytes)
  char* ws = (char*)d_ws;
  const size_t szHidH = (size_t)TT * HID * 2;      // 25.2 MB
  const size_t szQkvW = (size_t)QKVN * HID * 2;    // 22.0 MB
  const size_t szOW   = (size_t)HID * HID * 2;     // 18.9 MB
  const size_t szQkv  = (size_t)TT * QKVN * 2;     // 29.4 MB
  const size_t szQ    = (size_t)BB * HH * SS * DD * 2;
  const size_t szK    = (size_t)BB * KK * SS * DD * 2;
  const size_t szV    = (size_t)BB * KK * SS * DD * 2;
  size_t off = 0;
  _Float16* hidden_h = (_Float16*)(ws + off); off += szHidH;
  _Float16* qkvw_h   = (_Float16*)(ws + off); off += szQkvW;
  _Float16* ow_h     = (_Float16*)(ws + off); off += szOW;
  _Float16* qkv_h    = (_Float16*)(ws + off); off += szQkv;
  _Float16* q_h      = (_Float16*)(ws + off); off += szQ;
  _Float16* k_h      = (_Float16*)(ws + off); off += szK;
  _Float16* vT_h     = (_Float16*)(ws + off); off += szV;
  _Float16* attn_h   = (_Float16*)(ws + off); off += szHidH;

  // 1) f32 -> f16 conversions
  cvt_f16_kernel<<<4096, 256, 0, stream>>>(hidden, hidden_h, TT * HID);
  cvt_f16_kernel<<<4096, 256, 0, stream>>>(qkv_w, qkvw_h, QKVN * HID);
  cvt_f16_kernel<<<4096, 256, 0, stream>>>(o_w, ow_h, HID * HID);

  // 2) QKV GEMM: [4096,3072] x [3584,3072]^T -> f16 [4096,3584]
  {
    const int waves = (TT / 32) * (QKVN / 64);  // 128 * 56 = 7168
    gemm_wmma_kernel<1><<<waves / 8, 256, 0, stream>>>(
        hidden_h, qkvw_h, qkv_b, qkv_h, nullptr, TT, QKVN, HID);
  }

  // 3) RoPE + scatter into q/k/vT layouts
  {
    const int total = TT * (HH + 2 * KK) * 64;
    rope_scatter_kernel<<<(total + 255) / 256, 256, 0, stream>>>(
        qkv_h, cosT, sinT, q_h, k_h, vT_h);
  }

  // 4) Flash attention -> f16 [4096,3072]
  {
    const int waves = BB * HH * (SS / 16);  // 6144
    flash_attn_kernel<<<waves / 8, 256, 0, stream>>>(q_h, k_h, vT_h, attn_h);
  }

  // 5) Output GEMM: [4096,3072] x [3072,3072]^T + bias -> f32 d_out
  {
    const int waves = (TT / 32) * (HID / 64);  // 128 * 48 = 6144
    gemm_wmma_kernel<0><<<waves / 8, 256, 0, stream>>>(
        attn_h, ow_h, o_b, nullptr, out, TT, HID, HID);
  }

  (void)in_sizes; (void)n_in; (void)out_size; (void)ws_size;
}
